// MultiHeadCasualAttention_40707700032084
// MI455X (gfx1250) — compile-verified, matrix operand order checked by reading, not run on hardware
//
#include <hip/hip_runtime.h>

// ---------------------------------------------------------------------------
// MI455X (gfx1250) transformer block: LN -> QKV (WMMA bf16) -> attention
// (flash-style, WMMA, TDM-staged K/V) -> O-proj (+resid) -> LN -> MLP.
// All GEMM operands reach WMMA frags via TDM -> LDS -> ds_load_b128:
// weights are pre-transposed to [N][K] so A- and B-tiles share the 128x32
// row-major-over-K shape; V is written batched-transposed for P@V.
// ---------------------------------------------------------------------------

typedef __attribute__((ext_vector_type(16))) __bf16 v16bf;
typedef __attribute__((ext_vector_type(8)))  float  v8f;
typedef __attribute__((ext_vector_type(4)))  unsigned int v4u;
typedef __attribute__((ext_vector_type(4)))  int v4i;
typedef __attribute__((ext_vector_type(8)))  int v8i;
typedef unsigned short u16;
typedef unsigned int   u32;
typedef unsigned long long u64;

#define BD   4
#define SD   1024
#define DD   1024
#define HH   16
#define HDD  64
#define AHD  1024
#define MM   (BD * SD)   // 4096 rows

#if defined(__gfx1250__) && __has_builtin(__builtin_amdgcn_tensor_load_to_lds) && \
    __has_builtin(__builtin_amdgcn_s_wait_tensorcnt)
#define HAVE_TDM 1
#else
#define HAVE_TDM 0
#endif

struct FragU { u32 u[8]; };
struct W8    { u32 w[8]; };

static __device__ inline v16bf as_bf(FragU f) { return __builtin_bit_cast(v16bf, f); }

static __device__ inline u16 f2bf(float f) {
    u32 u = __builtin_bit_cast(u32, f);
    u32 r = u + 0x7FFFu + ((u >> 16) & 1u);   // round-to-nearest-even
    return (u16)(r >> 16);
}

// K-pair offset inside a 16x32 (MxK) bf16 A/B fragment, per ISA 7.12.2:
// vgpr v<4 -> K = half*8 + 2v ; vgpr v>=4 -> K = 16 + half*8 + 2(v-4)
static __device__ inline int kpair(int v, int half) {
    return ((v & 4) << 2) + half * 8 + ((v & 3) << 1);
}

#if HAVE_TDM
// Issue a TDM 2D tile load (bf16 elements) global -> LDS.
// D# packing per CDNA5 ISA (group0) / 8.4 (group1).
static __device__ inline void tdm_load_2d(const void* gaddr, u32 lds_off,
                                          u32 td0, u32 td1, u64 st0,
                                          u32 tl0, u32 tl1)
{
    const u64 ga = (u64)gaddr;
    v4u g0;
    g0[0] = 1u;                                           // count=1, user mode
    g0[1] = lds_off;                                      // lds_addr (bytes)
    g0[2] = (u32)(ga & 0xFFFFFFFFull);                    // global_addr[31:0]
    g0[3] = (u32)((ga >> 32) & 0x1FFFFFFull) | (2u << 30);// addr[56:32] | type=2
    W8 g;
    g.w[0] = 0x00010000u;                                 // data_size=1 (2 bytes)
    g.w[1] = (td0 & 0xFFFFu) << 16;                       // tensor_dim0[15:0]
    g.w[2] = (td0 >> 16) | ((td1 & 0xFFFFu) << 16);       // td0[31:16] | td1[15:0]
    g.w[3] = (td1 >> 16) | (tl0 << 16);                   // td1[31:16] | tile_dim0
    g.w[4] = tl1 & 0xFFFFu;                               // tile_dim1 (tile_dim2=0)
    g.w[5] = (u32)(st0 & 0xFFFFFFFFull);                  // dim0_stride[31:0]
    g.w[6] = (u32)((st0 >> 32) & 0xFFFFull);              // dim0_stride[47:32]
    g.w[7] = 0u;
    const v8i g1 = __builtin_bit_cast(v8i, g);
    const v4i z4 = (v4i)0;
#if __clang_major__ >= 23
    const v8i z8 = (v8i)0;
    __builtin_amdgcn_tensor_load_to_lds(g0, g1, z4, z4, z8, 0);
#else
    __builtin_amdgcn_tensor_load_to_lds(g0, g1, z4, z4, 0);
#endif
}
#endif

// ---------------------------------------------------------------------------
// Transpose + convert: in fp32 [Kd, Nd] -> out bf16 [Nd, Kd]. 32x32 tiles.
// ---------------------------------------------------------------------------
__global__ __launch_bounds__(256) void cvt_t_bf16_kernel(
    const float* __restrict__ in, u16* __restrict__ out, int Kd, int Nd)
{
    __shared__ float tile[32][33];
    const int n0 = blockIdx.x * 32, k0 = blockIdx.y * 32;
    const int tx = threadIdx.x & 31, ty = threadIdx.x >> 5;   // 32 x 8
    #pragma unroll
    for (int r = 0; r < 4; ++r)
        tile[ty + r * 8][tx] = in[(size_t)(k0 + ty + r * 8) * Nd + n0 + tx];
    __syncthreads();
    #pragma unroll
    for (int r = 0; r < 4; ++r)
        out[(size_t)(n0 + ty + r * 8) * Kd + k0 + tx] = f2bf(tile[tx][ty + r * 8]);
}

// ---------------------------------------------------------------------------
// LayerNorm (fp32 in, bf16 out). One block per row of Dn elements.
// ---------------------------------------------------------------------------
__global__ __launch_bounds__(256) void ln_bf16_kernel(
    const float* __restrict__ X, const float* __restrict__ sc,
    const float* __restrict__ bi, u16* __restrict__ Y, int Dn)
{
    const int row = blockIdx.x;
    const float* x = X + (size_t)row * Dn;
    float s1 = 0.f, s2 = 0.f;
    for (int i = threadIdx.x; i < Dn; i += 256) {
        float v = x[i];
        s1 += v; s2 += v * v;
    }
    #pragma unroll
    for (int off = 16; off; off >>= 1) {
        s1 += __shfl_xor(s1, off, 32);
        s2 += __shfl_xor(s2, off, 32);
    }
    __shared__ float w1[8], w2[8];
    if ((threadIdx.x & 31) == 0) { w1[threadIdx.x >> 5] = s1; w2[threadIdx.x >> 5] = s2; }
    __syncthreads();
    float t1 = 0.f, t2 = 0.f;
    #pragma unroll
    for (int i = 0; i < 8; ++i) { t1 += w1[i]; t2 += w2[i]; }
    const float mu  = t1 / (float)Dn;
    const float var = t2 / (float)Dn - mu * mu;
    const float inv = rsqrtf(var + 1e-5f);
    for (int i = threadIdx.x; i < Dn; i += 256) {
        float v = (x[i] - mu) * inv * sc[i] + bi[i];
        Y[(size_t)row * Dn + i] = f2bf(v);
    }
}

// ---------------------------------------------------------------------------
// bf16 WMMA GEMM, 128x128 block tile, K-step 32, TDM double-buffered LDS.
// A: [M,K] row-major. Bt: [N,K] row-major (transposed weights) so both tiles
// are 128x32 and every fragment is two ds_load_b128.
// 8 waves in 2x4; each wave computes 64x32 (4x2 WMMA tiles).
// Epilogue: +bias, optional sigmoid-form GELU, optional fp32 residual;
// fp32 out and/or bf16 out (tmode=1 stores bf16 batched-transposed for V).
// ---------------------------------------------------------------------------
__global__ __launch_bounds__(256) void gemm_bf16_kernel(
    const u16* __restrict__ A, const u16* __restrict__ Bt,
    const float* __restrict__ bias, const float* __restrict__ resid,
    float* __restrict__ outF, u16* __restrict__ outB,
    int M, int N, int K, int gelu, int tmode)
{
    const int lane  = threadIdx.x & 31;
    const int wave  = threadIdx.x >> 5;
    const int l15   = lane & 15;
    const int half  = lane >> 4;
    const int waveM = wave >> 2;       // 0..1
    const int waveN = wave & 3;        // 0..3
    const int m0 = blockIdx.y * 128;
    const int n0 = blockIdx.x * 128;

    __shared__ __attribute__((aligned(16))) u16 As[2][128 * 32];   // [m][k]
    __shared__ __attribute__((aligned(16))) u16 Bs[2][128 * 32];   // [n][k]

    v8f acc[4][2] = {};
    const int nk = K / 32;

#if HAVE_TDM
    if (wave == 0) {   // prologue: stage 0
        tdm_load_2d(A  + (size_t)m0 * K, (u32)(size_t)&As[0][0],
                    (u32)K, (u32)M, (u64)K, 32u, 128u);
        tdm_load_2d(Bt + (size_t)n0 * K, (u32)(size_t)&Bs[0][0],
                    (u32)K, (u32)N, (u64)K, 32u, 128u);
    }
#endif

    for (int i = 0; i < nk; ++i) {
        const int buf = i & 1;
#if HAVE_TDM
        if (wave == 0) {
            if (i + 1 < nk) {
                const int k1 = (i + 1) * 32;
                const int nb = (i + 1) & 1;
                tdm_load_2d(A  + (size_t)m0 * K + k1, (u32)(size_t)&As[nb][0],
                            (u32)K, (u32)M, (u64)K, 32u, 128u);
                tdm_load_2d(Bt + (size_t)n0 * K + k1, (u32)(size_t)&Bs[nb][0],
                            (u32)K, (u32)N, (u64)K, 32u, 128u);
                __builtin_amdgcn_s_wait_tensorcnt(2);   // stage i complete
            } else {
                __builtin_amdgcn_s_wait_tensorcnt(0);
            }
        }
        __syncthreads();
#else
        const int k0 = i * 32;
        for (int idx = threadIdx.x; idx < 512; idx += 256) {      // 128 rows x 64B
            const int r = idx >> 2, s = idx & 3;
            *(uint4*)&As[buf][r * 32 + s * 8] =
                *(const uint4*)(A + (size_t)(m0 + r) * K + k0 + s * 8);
            *(uint4*)&Bs[buf][r * 32 + s * 8] =
                *(const uint4*)(Bt + (size_t)(n0 + r) * K + k0 + s * 8);
        }
        __syncthreads();
#endif
        FragU fa[4], fb[2];
        #pragma unroll
        for (int t = 0; t < 4; ++t)
            #pragma unroll
            for (int v = 0; v < 8; ++v)
                fa[t].u[v] = *(const u32*)&As[buf][(waveM * 64 + t * 16 + l15) * 32 +
                                                   kpair(v, half)];
        #pragma unroll
        for (int u = 0; u < 2; ++u)
            #pragma unroll
            for (int v = 0; v < 8; ++v)
                fb[u].u[v] = *(const u32*)&Bs[buf][(waveN * 32 + u * 16 + l15) * 32 +
                                                   kpair(v, half)];
        #pragma unroll
        for (int t = 0; t < 4; ++t)
            #pragma unroll
            for (int u = 0; u < 2; ++u)
                acc[t][u] = __builtin_amdgcn_wmma_f32_16x16x32_bf16(
                    false, as_bf(fa[t]), false, as_bf(fb[u]), (short)0,
                    acc[t][u], false, false);
        __syncthreads();
    }

    #pragma unroll
    for (int t = 0; t < 4; ++t)
        #pragma unroll
        for (int u = 0; u < 2; ++u)
            #pragma unroll
            for (int r = 0; r < 8; ++r) {
                const int m = m0 + waveM * 64 + t * 16 + r + half * 8;
                const int n = n0 + waveN * 32 + u * 16 + l15;
                float c = acc[t][u][r];
                if (bias) c += bias[n];
                if (gelu) {   // 0.5x(1+tanh z) == x * sigmoid(2z)
                    const float xx = c;
                    c = xx / (1.f + __expf(-1.5957691216057308f *
                                           (xx + 0.044715f * xx * xx * xx)));
                }
                if (resid) c += resid[(size_t)m * N + n];
                if (outF)  outF[(size_t)m * N + n] = c;
                if (outB) {
                    if (tmode) {   // batched transpose: vt[(b*AHD+n)*S + s]
                        const int bb = m >> 10;       // m / SD
                        const int s  = m & (SD - 1);  // m % SD
                        outB[((size_t)bb * AHD + n) * SD + s] = f2bf(c);
                    } else {
                        outB[(size_t)m * N + n] = f2bf(c);
                    }
                }
            }
}

// ---------------------------------------------------------------------------
// Attention: block = 4 waves, one (b,h), 64 queries (16 per wave). K tile
// (32x64) and Vt tile (64x32) are shared by all 4 waves, staged in LDS via
// TDM double buffer. Online softmax over 32-key steps; P transposed through
// LDS (C-layout -> A-layout). Mask faithful to triu(-999, k=S-1).
// Vt is batched-transposed: vt[(b*AHD + n)*S + j].
// ---------------------------------------------------------------------------
__global__ __launch_bounds__(128) void attn_kernel(
    const u16* __restrict__ Q, const u16* __restrict__ Kb,
    const u16* __restrict__ Vt, u16* __restrict__ O)
{
    const int lane = threadIdx.x & 31;
    const int wave = threadIdx.x >> 5;
    const int l15  = lane & 15;
    const int half = lane >> 4;
    const int bh = blockIdx.y;
    const int b  = bh / HH, h = bh % HH;
    const int q0 = blockIdx.x * 64 + wave * 16;

    __shared__ __attribute__((aligned(16))) u16 Ks[2][32 * 64];   // [j][d]
    __shared__ __attribute__((aligned(16))) u16 Vs[2][64 * 32];   // [d][j]
    __shared__ __attribute__((aligned(16))) u16 pbuf[4][16 * 32];

    const u16* Kbase = Kb + (size_t)(b * SD) * AHD + h * HDD;     // + j*AHD + d
    const u16* Vbase = Vt + (size_t)(b * AHD + h * HDD) * SD;     // + d*SD + j

    const u16* Qrow = Q + (size_t)(b * SD + q0 + l15) * AHD + h * HDD;
    FragU fq[2];
    #pragma unroll
    for (int t = 0; t < 2; ++t)
        #pragma unroll
        for (int v = 0; v < 8; ++v)
            fq[t].u[v] = *(const u32*)(Qrow + t * 32 + kpair(v, half));

    v8f o[4] = {};
    float mrow[8], lrow[8];
    #pragma unroll
    for (int r = 0; r < 8; ++r) { mrow[r] = -1e30f; lrow[r] = 0.f; }
    const float scale = 1.0f / 32.0f;   // 1/sqrt(D), D=1024

#if HAVE_TDM
    if (wave == 0) {   // prologue: stage 0
        tdm_load_2d(Kbase, (u32)(size_t)&Ks[0][0], (u32)AHD, (u32)SD, (u64)AHD, 64u, 32u);
        tdm_load_2d(Vbase, (u32)(size_t)&Vs[0][0], (u32)SD, (u32)AHD, (u64)SD, 32u, 64u);
    }
#endif

    const int nj = SD / 32;
    for (int i = 0; i < nj; ++i) {
        const int j0  = i * 32;
        const int buf = i & 1;
#if HAVE_TDM
        if (wave == 0) {
            if (i + 1 < nj) {
                const int j1 = (i + 1) * 32;
                const int nb = (i + 1) & 1;
                tdm_load_2d(Kbase + (size_t)j1 * AHD, (u32)(size_t)&Ks[nb][0],
                            (u32)AHD, (u32)SD, (u64)AHD, 64u, 32u);
                tdm_load_2d(Vbase + j1, (u32)(size_t)&Vs[nb][0],
                            (u32)SD, (u32)AHD, (u64)SD, 32u, 64u);
                __builtin_amdgcn_s_wait_tensorcnt(2);
            } else {
                __builtin_amdgcn_s_wait_tensorcnt(0);
            }
        }
        __syncthreads();
#else
        for (int idx = threadIdx.x; idx < 256; idx += 128) {   // K: 32 rows x 128B
            const int r = idx >> 3, s = idx & 7;
            *(uint4*)&Ks[buf][r * 64 + s * 8] =
                *(const uint4*)(Kbase + (size_t)(j0 + r) * AHD + s * 8);
        }
        for (int idx = threadIdx.x; idx < 256; idx += 128) {   // Vt: 64 rows x 64B
            const int r = idx >> 2, s = idx & 3;
            *(uint4*)&Vs[buf][r * 32 + s * 8] =
                *(const uint4*)(Vbase + (size_t)r * SD + j0 + s * 8);
        }
        __syncthreads();
#endif
        // ---- scores: two 16x16 tiles over keys (B-frag from Ks, K-dim = d)
        v8f sc[2] = {};
        #pragma unroll
        for (int t = 0; t < 2; ++t)
            #pragma unroll
            for (int ks = 0; ks < 2; ++ks) {
                FragU fb;
                #pragma unroll
                for (int v = 0; v < 8; ++v)
                    fb.u[v] = *(const u32*)&Ks[buf][(t * 16 + l15) * 64 + ks * 32 +
                                                    kpair(v, half)];
                sc[t] = __builtin_amdgcn_wmma_f32_16x16x32_bf16(
                    false, as_bf(fq[ks]), false, as_bf(fb), (short)0, sc[t], false, false);
            }
        // ---- scale + mask + online softmax update (C-layout rows)
        #pragma unroll
        for (int r = 0; r < 8; ++r) {
            const int qg = q0 + r + half * 8;
            float s0 = sc[0][r] * scale;
            float s1 = sc[1][r] * scale;
            if ((j0      + l15) - qg >= SD - 1) s0 -= 999.f;
            if ((j0 + 16 + l15) - qg >= SD - 1) s1 -= 999.f;
            float mx = fmaxf(s0, s1);
            #pragma unroll
            for (int off = 8; off; off >>= 1) mx = fmaxf(mx, __shfl_xor(mx, off, 32));
            const float mnew  = fmaxf(mrow[r], mx);
            const float alpha = __expf(mrow[r] - mnew);
            const float p0 = __expf(s0 - mnew);
            const float p1 = __expf(s1 - mnew);
            float ps = p0 + p1;
            #pragma unroll
            for (int off = 8; off; off >>= 1) ps += __shfl_xor(ps, off, 32);
            lrow[r] = lrow[r] * alpha + ps;
            mrow[r] = mnew;
            #pragma unroll
            for (int d = 0; d < 4; ++d) o[d][r] *= alpha;
            pbuf[wave][(r + half * 8) * 32 +      l15] = f2bf(p0);
            pbuf[wave][(r + half * 8) * 32 + 16 + l15] = f2bf(p1);
        }
        __syncthreads();
        // ---- P @ V : P A-frag from LDS, V B-frag from Vs (K-dim = j)
        FragU fp;
        #pragma unroll
        for (int v = 0; v < 8; ++v)
            fp.u[v] = *(const u32*)&pbuf[wave][l15 * 32 + kpair(v, half)];
        #pragma unroll
        for (int d = 0; d < 4; ++d) {
            FragU fv;
            #pragma unroll
            for (int v = 0; v < 8; ++v)
                fv.u[v] = *(const u32*)&Vs[buf][(d * 16 + l15) * 32 + kpair(v, half)];
            o[d] = __builtin_amdgcn_wmma_f32_16x16x32_bf16(
                false, as_bf(fp), false, as_bf(fv), (short)0, o[d], false, false);
        }
        __syncthreads();
    }
    #pragma unroll
    for (int d = 0; d < 4; ++d)
        #pragma unroll
        for (int r = 0; r < 8; ++r) {
            const int m = q0 + r + half * 8;
            float val = o[d][r] / lrow[r];
            O[(size_t)(b * SD + m) * AHD + h * HDD + d * 16 + l15] = f2bf(val);
        }
}

// ---------------------------------------------------------------------------
extern "C" void kernel_launch(void* const* d_in, const int* in_sizes, int n_in,
                              void* d_out, int out_size, void* d_ws, size_t ws_size,
                              hipStream_t stream) {
    (void)in_sizes; (void)n_in; (void)out_size; (void)ws_size;
    const float* x     = (const float*)d_in[0];
    const float* lnq_s = (const float*)d_in[1];
    const float* lnq_b = (const float*)d_in[2];
    const float* Wq    = (const float*)d_in[3];
    const float* bq    = (const float*)d_in[4];
    const float* lnk_s = (const float*)d_in[5];
    const float* lnk_b = (const float*)d_in[6];
    const float* Wk    = (const float*)d_in[7];
    const float* bk    = (const float*)d_in[8];
    const float* lnv_s = (const float*)d_in[9];
    const float* lnv_b = (const float*)d_in[10];
    const float* Wv    = (const float*)d_in[11];
    const float* bv    = (const float*)d_in[12];
    const float* Wo    = (const float*)d_in[13];
    const float* bo    = (const float*)d_in[14];
    const float* lnm_s = (const float*)d_in[15];
    const float* lnm_b = (const float*)d_in[16];
    const float* W1    = (const float*)d_in[17];
    const float* b1    = (const float*)d_in[18];
    const float* W2    = (const float*)d_in[19];
    const float* b2    = (const float*)d_in[20];
    float* out = (float*)d_out;

    char* p = (char*)d_ws;
    auto carve = [&](size_t bytes) -> void* {
        void* r = (void*)p;
        p += (bytes + 255) & ~(size_t)255;
        return r;
    };
    u16* wqT   = (u16*)carve((size_t)DD * AHD * 2);       // [AHD][DD]
    u16* wkT   = (u16*)carve((size_t)DD * AHD * 2);
    u16* wvT   = (u16*)carve((size_t)DD * AHD * 2);
    u16* woT   = (u16*)carve((size_t)AHD * DD * 2);       // [DD][AHD]
    u16* w1T   = (u16*)carve((size_t)DD * 2 * DD * 2);    // [2DD][DD]
    u16* w2T   = (u16*)carve((size_t)2 * DD * DD * 2);    // [DD][2DD]
    u16* lnq16 = (u16*)carve((size_t)MM * DD * 2);
    u16* lnk16 = (u16*)carve((size_t)MM * DD * 2);
    u16* lnv16 = (u16*)carve((size_t)MM * DD * 2);
    u16* q16   = (u16*)carve((size_t)MM * AHD * 2);
    u16* k16   = (u16*)carve((size_t)MM * AHD * 2);
    u16* vt16  = (u16*)carve((size_t)MM * AHD * 2);       // batched-transposed V
    u16* att16 = (u16*)carve((size_t)MM * AHD * 2);
    float* hbuf= (float*)carve((size_t)MM * DD * 4);
    u16* lnm16 = (u16*)carve((size_t)MM * DD * 2);
    u16* g16   = (u16*)carve((size_t)MM * 2 * DD * 2);

    // ---- weight transpose-conversions: W[K,N] -> Wt[N,K] bf16
    cvt_t_bf16_kernel<<<dim3(AHD / 32, DD / 32), 256, 0, stream>>>(Wq, wqT, DD, AHD);
    cvt_t_bf16_kernel<<<dim3(AHD / 32, DD / 32), 256, 0, stream>>>(Wk, wkT, DD, AHD);
    cvt_t_bf16_kernel<<<dim3(AHD / 32, DD / 32), 256, 0, stream>>>(Wv, wvT, DD, AHD);
    cvt_t_bf16_kernel<<<dim3(DD / 32, AHD / 32), 256, 0, stream>>>(Wo, woT, AHD, DD);
    cvt_t_bf16_kernel<<<dim3(2 * DD / 32, DD / 32), 256, 0, stream>>>(W1, w1T, DD, 2 * DD);
    cvt_t_bf16_kernel<<<dim3(DD / 32, 2 * DD / 32), 256, 0, stream>>>(W2, w2T, 2 * DD, DD);

    // ---- branch LayerNorms of x
    ln_bf16_kernel<<<MM, 256, 0, stream>>>(x, lnq_s, lnq_b, lnq16, DD);
    ln_bf16_kernel<<<MM, 256, 0, stream>>>(x, lnk_s, lnk_b, lnk16, DD);
    ln_bf16_kernel<<<MM, 256, 0, stream>>>(x, lnv_s, lnv_b, lnv16, DD);

    // ---- Q/K/V projections (V stored batched-transposed)
    dim3 gQKV(AHD / 128, MM / 128);
    gemm_bf16_kernel<<<gQKV, 256, 0, stream>>>(lnq16, wqT, bq, nullptr, nullptr, q16,
                                               MM, AHD, DD, 0, 0);
    gemm_bf16_kernel<<<gQKV, 256, 0, stream>>>(lnk16, wkT, bk, nullptr, nullptr, k16,
                                               MM, AHD, DD, 0, 0);
    gemm_bf16_kernel<<<gQKV, 256, 0, stream>>>(lnv16, wvT, bv, nullptr, nullptr, vt16,
                                               MM, AHD, DD, 0, 1);

    // ---- attention
    dim3 gAtt(SD / 64, BD * HH);
    attn_kernel<<<gAtt, 128, 0, stream>>>(q16, k16, vt16, att16);

    // ---- output projection + bias + residual -> fp32 h
    dim3 gO(DD / 128, MM / 128);
    gemm_bf16_kernel<<<gO, 256, 0, stream>>>(att16, woT, bo, x, hbuf, nullptr,
                                             MM, DD, AHD, 0, 0);

    // ---- MLP: LN(h) -> GEMM+GELU -> GEMM + bias + residual(h) -> out
    ln_bf16_kernel<<<MM, 256, 0, stream>>>(hbuf, lnm_s, lnm_b, lnm16, DD);
    dim3 g1(2 * DD / 128, MM / 128);
    gemm_bf16_kernel<<<g1, 256, 0, stream>>>(lnm16, w1T, b1, nullptr, nullptr, g16,
                                             MM, 2 * DD, DD, 1, 0);
    dim3 g2(DD / 128, MM / 128);
    gemm_bf16_kernel<<<g2, 256, 0, stream>>>(g16, w2T, b2, hbuf, out, nullptr,
                                             MM, DD, 2 * DD, 0, 0);
}
